// LSHAttention_21474836480605
// MI455X (gfx1250) — matrix operand order repaired and checked by reading, not compiled, not run on hardware
//
#include <hip/hip_runtime.h>

// ---------------------------------------------------------------------------
// LSH attention for MI455X (gfx1250): bf16 WMMA everywhere, flash softmax.
// B=2, L=2048, F=1024, H=16, HD=64, N_HASHES=4
// ---------------------------------------------------------------------------

#define Bb 2
#define Ll 2048
#define Ff 1024
#define Hh 16
#define HD 64
#define BH (Bb*Hh)
#define Mtot (Bb*Ll)   // 4096
#define KTC  32        // K/32 for all GEMMs here (K = 1024)

typedef __attribute__((ext_vector_type(16))) __bf16 v16bf;
typedef __attribute__((ext_vector_type(8)))  float  v8f;

union Frag { uint4 u[2]; v16bf h; };

__device__ __forceinline__ unsigned short f2bf(float f) {
    unsigned u = __builtin_bit_cast(unsigned, f);
    unsigned r = u + 0x7FFFu + ((u >> 16) & 1u);   // round-to-nearest-even
    return (unsigned short)(r >> 16);
}

// ---------------------------------------------------------------------------
// Pack fp32 [M,K] row-major -> WMMA A-fragment bf16 layout:
//   [M/16][K/32][lane 0..31][v 0..7] uints (uint = bf16 pair)
//   lane: m = lane&15, khalf = lane>>4 ; v: k = (v>>2)*16 + khalf*8 + (v&3)*2
// ---------------------------------------------------------------------------
__global__ void pack_a_frag(const float* __restrict__ src, unsigned* __restrict__ dst,
                            int M, int K) {
    size_t t = (size_t)blockIdx.x * blockDim.x + threadIdx.x;
    size_t total = (size_t)M * K / 2;
    if (t >= total) return;
    int v    = (int)(t & 7);
    int lane = (int)((t >> 3) & 31);
    size_t tile = t >> 8;
    int ktc = K / 32;
    int kt = (int)(tile % ktc);
    int mt = (int)(tile / ktc);
    int m = mt * 16 + (lane & 15);
    int k = kt * 32 + ((v >> 2) * 16) + ((lane >> 4) * 8) + ((v & 3) * 2);
    const float* p = src + (size_t)m * K + k;
    dst[t] = (unsigned)f2bf(p[0]) | ((unsigned)f2bf(p[1]) << 16);
}

// ---------------------------------------------------------------------------
// Pack fp32 W [K,N] row-major -> WMMA B-fragment bf16 layout:
//   [N/16][K/32][lane][v] ; lane: n = lane&15, khalf = lane>>4 ; k = (v>>2)*16 + khalf*8 + (v&3)*2
// ---------------------------------------------------------------------------
__global__ void pack_b_frag(const float* __restrict__ W, unsigned* __restrict__ dst,
                            int K, int N) {
    size_t t = (size_t)blockIdx.x * blockDim.x + threadIdx.x;
    size_t total = (size_t)K * N / 2;
    if (t >= total) return;
    int v    = (int)(t & 7);
    int lane = (int)((t >> 3) & 31);
    size_t tile = t >> 8;
    int ktc = K / 32;
    int kt = (int)(tile % ktc);
    int nt = (int)(tile / ktc);
    int n = nt * 16 + (lane & 15);
    int k = kt * 32 + ((v >> 2) * 16) + ((lane >> 4) * 8) + ((v & 3) * 2);
    float e0 = W[(size_t)k * N + n];
    float e1 = W[(size_t)(k + 1) * N + n];
    dst[t] = (unsigned)f2bf(e0) | ((unsigned)f2bf(e1) << 16);
}

// ---------------------------------------------------------------------------
// WMMA GEMM (K fixed = 1024): C[M,N] = A_pack @ B_pack + bias (fp32 out)
// block = 128 threads = 4 waves; wave w -> 16x64 tile:
//   mt = by*4+w, n-tiles bx*4 .. bx*4+3 (A-frag reused by 4 WMMAs per k-step)
// Single base pointer per stream; B n-tiles at constant 32 KB immediate
// offsets -> pure global_load_b128 + partial s_wait_loadcnt pipelining.
// ---------------------------------------------------------------------------
__global__ __launch_bounds__(128) void gemm_wmma_bf16(
    const uint4* __restrict__ Ap, const uint4* __restrict__ Bp,
    const float* __restrict__ bias, float* __restrict__ C,
    int M, int N) {
    int lane = threadIdx.x & 31;
    int w    = threadIdx.x >> 5;
    int nt0 = blockIdx.x * 4;
    int mt  = blockIdx.y * 4 + w;
    const uint4* a = Ap + ((size_t)mt  * KTC * 32 + lane) * 2;
    const uint4* b = Bp + ((size_t)nt0 * KTC * 32 + lane) * 2;
    const v8f vz = {0.f,0.f,0.f,0.f,0.f,0.f,0.f,0.f};
    v8f c0 = vz, c1 = vz, c2 = vz, c3 = vz;
#pragma unroll
    for (int kt = 0; kt < KTC; ++kt) {
        Frag af;
        af.u[0] = a[0]; af.u[1] = a[1];
        __builtin_prefetch(a + 128, 0, 1);   // global_prefetch_b8
        a += 64;                              // next k-tile (32 lanes * 2 uint4)
        Frag bf0, bf1, bf2, bf3;
        bf0.u[0] = b[0 * KTC * 64 + 0]; bf0.u[1] = b[0 * KTC * 64 + 1];
        bf1.u[0] = b[1 * KTC * 64 + 0]; bf1.u[1] = b[1 * KTC * 64 + 1];
        bf2.u[0] = b[2 * KTC * 64 + 0]; bf2.u[1] = b[2 * KTC * 64 + 1];
        bf3.u[0] = b[3 * KTC * 64 + 0]; bf3.u[1] = b[3 * KTC * 64 + 1];
        b += 64;
        c0 = __builtin_amdgcn_wmma_f32_16x16x32_bf16(false, af.h, false, bf0.h,
                                                     (short)0, c0, false, false);
        c1 = __builtin_amdgcn_wmma_f32_16x16x32_bf16(false, af.h, false, bf1.h,
                                                     (short)0, c1, false, false);
        c2 = __builtin_amdgcn_wmma_f32_16x16x32_bf16(false, af.h, false, bf2.h,
                                                     (short)0, c2, false, false);
        c3 = __builtin_amdgcn_wmma_f32_16x16x32_bf16(false, af.h, false, bf3.h,
                                                     (short)0, c3, false, false);
    }
    int mbase = mt * 16 + ((lane >> 4) * 8);
    v8f cacc[4] = {c0, c1, c2, c3};
#pragma unroll
    for (int j = 0; j < 4; ++j) {
        int n = (nt0 + j) * 16 + (lane & 15);
        float bv = bias ? bias[n] : 0.f;
#pragma unroll
        for (int r = 0; r < 8; ++r)
            C[(size_t)(mbase + r) * N + n] = cacc[j][r] + bv;
    }
}

// ---------------------------------------------------------------------------
// LSH bucket assignment: rotated = q . R ; bucket = argmax([rot, -rot])
// qk: [B,L,2F] fp32 ; R: [H,4,64] ; buckets: [B,H,L] uint
// ---------------------------------------------------------------------------
__global__ void lsh_buckets(const float* __restrict__ qk, const float* __restrict__ R,
                            unsigned* __restrict__ buckets) {
    int t = blockIdx.x * blockDim.x + threadIdx.x;
    if (t >= Bb * Hh * Ll) return;
    int l = t & (Ll - 1);
    int h = (t >> 11) & (Hh - 1);
    int b = t >> 15;
    const float* q  = qk + ((size_t)(b * Ll + l)) * (2 * Ff) + h * HD;
    const float* Rh = R + (size_t)h * 4 * HD;
    float d0 = 0.f, d1 = 0.f, d2 = 0.f, d3 = 0.f;
#pragma unroll 4
    for (int i = 0; i < HD; ++i) {
        float qv = q[i];
        d0 += qv * Rh[0 * HD + i];
        d1 += qv * Rh[1 * HD + i];
        d2 += qv * Rh[2 * HD + i];
        d3 += qv * Rh[3 * HD + i];
    }
    float dv[8] = {d0, d1, d2, d3, -d0, -d1, -d2, -d3};
    int best = 0; float bvv = dv[0];
#pragma unroll
    for (int r = 1; r < 8; ++r) if (dv[r] > bvv) { bvv = dv[r]; best = r; }
    buckets[t] = (unsigned)best;
}

// ---------------------------------------------------------------------------
// Stable counting sort of 2048 positions by bucket (8 bins), one wave32 per
// (b,h). Uses __ballot masks for a stable parallel rank. idx: [BH,L] int.
// ---------------------------------------------------------------------------
__global__ __launch_bounds__(32) void lsh_sort(const unsigned* __restrict__ buckets,
                                               int* __restrict__ idx) {
    int bh = blockIdx.x;
    int lane = threadIdx.x;
    const unsigned* bk = buckets + (size_t)bh * Ll;
    int cnt[8];
#pragma unroll
    for (int j = 0; j < 8; ++j) cnt[j] = 0;
    for (int c = 0; c < Ll / 32; ++c) {
        unsigned bb = bk[c * 32 + lane];
#pragma unroll
        for (int j = 0; j < 8; ++j) {
            unsigned long long m = __ballot(bb == (unsigned)j);
            cnt[j] += __popcll(m);
        }
    }
    int run[8]; int acc = 0;
#pragma unroll
    for (int j = 0; j < 8; ++j) { run[j] = acc; acc += cnt[j]; }
    unsigned lmask = (1u << lane) - 1u;
    for (int c = 0; c < Ll / 32; ++c) {
        unsigned bb = bk[c * 32 + lane];
        unsigned mm[8];
#pragma unroll
        for (int j = 0; j < 8; ++j)
            mm[j] = (unsigned)__ballot(bb == (unsigned)j);
#pragma unroll
        for (int j = 0; j < 8; ++j)
            if (bb == (unsigned)j) {
                int pos = run[j] + __popc(mm[j] & lmask);
                idx[(size_t)bh * Ll + pos] = c * 32 + lane;
            }
#pragma unroll
        for (int j = 0; j < 8; ++j) run[j] += __popc(mm[j]);
    }
}

// ---------------------------------------------------------------------------
// Gather sorted q,k,v into bf16: sq [BH,L,64] (scale 1/8 folded),
// sk [BH,L,64], svT [BH,64,L] (transposed for contiguous PV B-fragments)
// ---------------------------------------------------------------------------
__global__ void gather_sorted(const float* __restrict__ qk, const float* __restrict__ vbuf,
                              const int* __restrict__ idx,
                              unsigned short* __restrict__ sq,
                              unsigned short* __restrict__ sk,
                              unsigned short* __restrict__ svT) {
    size_t t = (size_t)blockIdx.x * blockDim.x + threadIdx.x;
    if (t >= (size_t)BH * Ll * HD) return;
    int d  = (int)(t & (HD - 1));
    int r  = (int)((t >> 6) & (Ll - 1));
    int bh = (int)(t >> 17);
    int b = bh >> 4, h = bh & (Hh - 1);
    int l = idx[(size_t)bh * Ll + r];
    size_t qrow = ((size_t)(b * Ll + l)) * (2 * Ff) + h * HD + d;
    float qv = qk[qrow] * 0.125f;            // 1/sqrt(64)
    float kv = qk[qrow + Ff];
    float vv = vbuf[((size_t)(b * Ll + l)) * Ff + h * HD + d];
    sq[((size_t)bh * Ll + r) * HD + d] = f2bf(qv);
    sk[((size_t)bh * Ll + r) * HD + d] = f2bf(kv);
    svT[((size_t)bh * HD + d) * Ll + r] = f2bf(vv);
}

// ---------------------------------------------------------------------------
// Flash attention over sorted sequence, WMMA bf16, online softmax.
// block = 256 (8 waves); wave -> 16 query rows; loop keys in steps of 32.
// K/V tiles staged in LDS once per block and shared by all 8 waves
// (8x reduction in L2 traffic vs per-wave loads -> WMMA-roofline bound).
// Output scattered straight back to original order: out [B,L,F] fp32.
// ---------------------------------------------------------------------------
__global__ __launch_bounds__(256) void flash_attn_sorted(
    const unsigned short* __restrict__ sq, const unsigned short* __restrict__ sk,
    const unsigned short* __restrict__ svT, const int* __restrict__ idx,
    float* __restrict__ out) {
    __shared__ __align__(16) unsigned short pb[8][16][32];   // P staging (per wave)
    __shared__ __align__(16) unsigned short kt_s[32][64];    // K tile  [key][d]
    __shared__ __align__(16) unsigned short vt_s[64][32];    // V tile  [d][key]
    int tid  = threadIdx.x;
    int lane = tid & 31;
    int w    = tid >> 5;
    int bh = blockIdx.y;
    int m0 = blockIdx.x * 128 + w * 16;
    int ml = lane & 15;
    int kh = lane >> 4;

    // Q fragments (16x64 = two 16x32 A-frags), loaded once
    const unsigned short* qbase = sq + ((size_t)bh * Ll + m0 + ml) * HD;
    Frag qf[2];
#pragma unroll
    for (int kk = 0; kk < 2; ++kk) {
        qf[kk].u[0] = *(const uint4*)(qbase + kk * 32 + kh * 16);
        qf[kk].u[1] = *(const uint4*)(qbase + kk * 32 + kh * 16 + 8);
    }

    float mprev[8], lsum[8];
    v8f o[4];
    const v8f vz = {0.f,0.f,0.f,0.f,0.f,0.f,0.f,0.f};
#pragma unroll
    for (int r = 0; r < 8; ++r) { mprev[r] = -3.0e38f; lsum[r] = 0.f; }
#pragma unroll
    for (int t = 0; t < 4; ++t) o[t] = vz;

    const unsigned short* kb = sk  + (size_t)bh * Ll * HD;
    const unsigned short* vb = svT + (size_t)bh * HD * Ll;

    // cooperative staging addresses (1 uint4 per thread per tile)
    int krow = tid >> 3, kcol = tid & 7;    // K: 32 rows x 8 uint4
    int vrow = tid >> 2, vcol = tid & 3;    // V: 64 rows x 4 uint4

    for (int n0 = 0; n0 < Ll; n0 += 32) {
        // ---- stage K/V tiles into LDS (shared by 8 waves) ----
        *(uint4*)&kt_s[krow][kcol * 8] =
            *(const uint4*)(kb + (size_t)(n0 + krow) * HD + kcol * 8);
        *(uint4*)&vt_s[vrow][vcol * 8] =
            *(const uint4*)(vb + (size_t)vrow * Ll + n0 + vcol * 8);
        __syncthreads();

        // ---- S = Q K^T for 32 keys: two 16x16 C tiles ----
        v8f c0 = vz, c1 = vz;
#pragma unroll
        for (int kk = 0; kk < 2; ++kk) {
            Frag kf0, kf1;
            const unsigned short* kp0 = &kt_s[ml][kk * 32 + kh * 16];
            kf0.u[0] = *(const uint4*)kp0;
            kf0.u[1] = *(const uint4*)(kp0 + 8);
            const unsigned short* kp1 = &kt_s[16 + ml][kk * 32 + kh * 16];
            kf1.u[0] = *(const uint4*)kp1;
            kf1.u[1] = *(const uint4*)(kp1 + 8);
            c0 = __builtin_amdgcn_wmma_f32_16x16x32_bf16(false, qf[kk].h, false, kf0.h,
                                                         (short)0, c0, false, false);
            c1 = __builtin_amdgcn_wmma_f32_16x16x32_bf16(false, qf[kk].h, false, kf1.h,
                                                         (short)0, c1, false, false);
        }

        // ---- online softmax (rows live across 16-lane groups) ----
#pragma unroll
        for (int r = 0; r < 8; ++r) {
            float s0 = c0[r], s1 = c1[r];
            float mx = fmaxf(s0, s1);
#pragma unroll
            for (int off = 1; off < 16; off <<= 1)
                mx = fmaxf(mx, __shfl_xor(mx, off, 32));
            float mnew = fmaxf(mprev[r], mx);
            float corr = __expf(mprev[r] - mnew);
            float p0 = __expf(s0 - mnew);
            float p1 = __expf(s1 - mnew);
            float ps = p0 + p1;
#pragma unroll
            for (int off = 1; off < 16; off <<= 1)
                ps += __shfl_xor(ps, off, 32);
            lsum[r] = lsum[r] * corr + ps;
            mprev[r] = mnew;
            o[0][r] *= corr; o[1][r] *= corr; o[2][r] *= corr; o[3][r] *= corr;
            int row = r + kh * 8;
            pb[w][row][ml]      = f2bf(p0);
            pb[w][row][16 + ml] = f2bf(p1);
        }

        // ---- re-layout P (C-layout -> A-fragment) through LDS (wave-local) ----
        Frag pf;
        pf.u[0] = *(const uint4*)&pb[w][ml][kh * 16];
        pf.u[1] = *(const uint4*)&pb[w][ml][kh * 16 + 8];

        // ---- O += P @ V  (V fragments contiguous thanks to [d][key] tile) ----
#pragma unroll
        for (int t = 0; t < 4; ++t) {
            const unsigned short* vp = &vt_s[t * 16 + ml][kh * 16];
            Frag vf;
            vf.u[0] = *(const uint4*)vp;
            vf.u[1] = *(const uint4*)(vp + 8);
            o[t] = __builtin_amdgcn_wmma_f32_16x16x32_bf16(false, pf.h, false, vf.h,
                                                           (short)0, o[t], false, false);
        }
        __syncthreads();   // protect kt_s/vt_s before next staging
    }

    // ---- normalize + scatter back to original token order ----
    int b = bh >> 4, h = bh & (Hh - 1);
#pragma unroll
    for (int r = 0; r < 8; ++r) {
        float inv = 1.f / lsum[r];
        int m = m0 + r + kh * 8;
        int lorig = idx[(size_t)bh * Ll + m];
        float* op = out + ((size_t)(b * Ll + lorig)) * Ff + h * HD + ml;
#pragma unroll
        for (int t = 0; t < 4; ++t) op[t * 16] = o[t][r] * inv;
    }
}

// ---------------------------------------------------------------------------
// Host-side launch
// ---------------------------------------------------------------------------
extern "C" void kernel_launch(void* const* d_in, const int* in_sizes, int n_in,
                              void* d_out, int out_size, void* d_ws, size_t ws_size,
                              hipStream_t stream) {
    const float* x   = (const float*)d_in[0];
    const float* Wqk = (const float*)d_in[1];
    const float* bqk = (const float*)d_in[2];
    const float* Wv  = (const float*)d_in[3];
    const float* bv  = (const float*)d_in[4];
    const float* Wo  = (const float*)d_in[5];
    const float* bo  = (const float*)d_in[6];
    const float* R   = (const float*)d_in[7];
    float* out = (float*)d_out;

    char* p = (char*)d_ws;
    auto alloc = [&](size_t bytes) -> char* {
        char* r = p;
        p += (bytes + 255) & ~(size_t)255;
        return r;
    };

    unsigned* xAp    = (unsigned*)alloc((size_t)Mtot * Ff * 2);          // 8 MB
    unsigned* Wqkp   = (unsigned*)alloc((size_t)Ff * 2 * Ff * 2);        // 4 MB
    unsigned* Wvp    = (unsigned*)alloc((size_t)Ff * Ff * 2);            // 2 MB
    unsigned* Wop    = (unsigned*)alloc((size_t)Ff * Ff * 2);            // 2 MB
    float*    qkbuf  = (float*)alloc((size_t)Mtot * 2 * Ff * 4);         // 32 MB
    float*    vbuf   = (float*)alloc((size_t)Mtot * Ff * 4);             // 16 MB
    unsigned* buckets= (unsigned*)alloc((size_t)BH * Ll * 4);
    int*      idx    = (int*)alloc((size_t)BH * Ll * 4);
    unsigned short* sq  = (unsigned short*)alloc((size_t)BH * Ll * HD * 2);
    unsigned short* sk  = (unsigned short*)alloc((size_t)BH * Ll * HD * 2);
    unsigned short* svT = (unsigned short*)alloc((size_t)BH * Ll * HD * 2);
    float*    attnout= (float*)alloc((size_t)Mtot * Ff * 4);             // 16 MB
    unsigned* attnAp = (unsigned*)alloc((size_t)Mtot * Ff * 2);          // 8 MB
    (void)ws_size; (void)n_in; (void)in_sizes; (void)out_size;

    // 1. pack activations & weights to WMMA fragment layouts
    {
        size_t n = (size_t)Mtot * Ff / 2;
        pack_a_frag<<<dim3((unsigned)((n + 255) / 256)), 256, 0, stream>>>(x, xAp, Mtot, Ff);
    }
    {
        size_t n = (size_t)Ff * 2 * Ff / 2;
        pack_b_frag<<<dim3((unsigned)((n + 255) / 256)), 256, 0, stream>>>(Wqk, Wqkp, Ff, 2 * Ff);
    }
    {
        size_t n = (size_t)Ff * Ff / 2;
        pack_b_frag<<<dim3((unsigned)((n + 255) / 256)), 256, 0, stream>>>(Wv, Wvp, Ff, Ff);
        pack_b_frag<<<dim3((unsigned)((n + 255) / 256)), 256, 0, stream>>>(Wo, Wop, Ff, Ff);
    }

    // 2. projections: qk = x@Wqk + bqk ; v = x@Wv + bv
    gemm_wmma_bf16<<<dim3((2 * Ff) / 64, Mtot / 64), 128, 0, stream>>>(
        (const uint4*)xAp, (const uint4*)Wqkp, bqk, qkbuf, Mtot, 2 * Ff);
    gemm_wmma_bf16<<<dim3(Ff / 64, Mtot / 64), 128, 0, stream>>>(
        (const uint4*)xAp, (const uint4*)Wvp, bv, vbuf, Mtot, Ff);

    // 3. LSH buckets
    lsh_buckets<<<dim3((Bb * Hh * Ll + 255) / 256), 256, 0, stream>>>(qkbuf, R, buckets);

    // 4. stable counting sort per (b,h): one wave32 per group
    lsh_sort<<<dim3(BH), 32, 0, stream>>>(buckets, idx);

    // 5. gather sorted q,k,v (bf16; q pre-scaled; v transposed)
    {
        size_t n = (size_t)BH * Ll * HD;
        gather_sorted<<<dim3((unsigned)((n + 255) / 256)), 256, 0, stream>>>(
            qkbuf, vbuf, idx, sq, sk, svT);
    }

    // 6. flash attention over sorted sequence, scatter to original order
    flash_attn_sorted<<<dim3(Ll / 128, BH), 256, 0, stream>>>(sq, sk, svT, idx, attnout);

    // 7. output projection: out = attnout @ Wo + bo
    {
        size_t n = (size_t)Mtot * Ff / 2;
        pack_a_frag<<<dim3((unsigned)((n + 255) / 256)), 256, 0, stream>>>(attnout, attnAp, Mtot, Ff);
    }
    gemm_wmma_bf16<<<dim3(Ff / 64, Mtot / 64), 128, 0, stream>>>(
        (const uint4*)attnAp, (const uint4*)Wop, bo, out, Mtot, Ff);
}
